// GAT_24885040513573
// MI455X (gfx1250) — compile-verified, matrix operand order checked by reading, not run on hardware
//
#include <hip/hip_runtime.h>

typedef _Float16 half_t;
typedef __attribute__((ext_vector_type(16))) _Float16 v16h;
typedef __attribute__((ext_vector_type(8)))  _Float16 v8h;
typedef __attribute__((ext_vector_type(8)))  float    v8f;

// ---- problem constants (from reference) ----
constexpr int NN   = 33;            // nodes per graph
constexpr int DEG  = 8;
constexpr int EG   = NN * DEG;      // 264 explicit edges per graph
constexpr int NG   = 2048;          // graphs
constexpr int NTOT = NN * NG;       // 67584 nodes
constexpr int ETOT = NG * EG;       // 540672 explicit edges total
constexpr float NEG_SLOPE = 0.2f;

// ---- ordered-uint encoding for float atomicMax in LDS ----
__device__ __forceinline__ unsigned f2ord(float f) {
    unsigned u = __float_as_uint(f);
    return (u & 0x80000000u) ? ~u : (u | 0x80000000u);
}
__device__ __forceinline__ float ord2f(unsigned u) {
    return (u & 0x80000000u) ? __uint_as_float(u & 0x7FFFFFFFu) : __uint_as_float(~u);
}
__device__ __forceinline__ float lrelu(float v) { return v > 0.f ? v : NEG_SLOPE * v; }
__device__ __forceinline__ float relu_(float v) { return v > 0.f ? v : 0.f; }

// =====================================================================
// prep: per-head dots ws[h]=W1_h·a_src1_h, wd[h]=W1_h·a_dst1_h  (8 floats)
// =====================================================================
__global__ void prep_alpha1_kernel(const float* __restrict__ W1,
                                   const float* __restrict__ a_src1,
                                   const float* __restrict__ a_dst1,
                                   float* __restrict__ wsd) {
    int t = threadIdx.x;
    if (t < 8) {
        int h = t & 3;
        const float* a = (t < 4) ? a_src1 : a_dst1;
        float s = 0.f;
        for (int c = 0; c < 64; ++c) s += W1[h * 64 + c] * a[h * 64 + c];
        wsd[t] = s;
    }
}

// transpose [K,N] f32 -> [N,K] f16 so WMMA B-fragments are contiguous in K
__global__ void transpose_f16_kernel(const float* __restrict__ W,
                                     half_t* __restrict__ WT, int K, int N) {
    int i = blockIdx.x * blockDim.x + threadIdx.x;
    if (i < K * N) {
        int k = i / N, n = i - k * N;
        WT[(size_t)n * K + k] = (half_t)W[i];
    }
}

// =====================================================================
// GAT layer 1 (in=1): fully factorized, one graph per block, LDS softmax
// =====================================================================
__global__ void gat1_kernel(const float* __restrict__ x,
                            const int* __restrict__ src,
                            const int* __restrict__ dst,
                            const float* __restrict__ W1,
                            const float* __restrict__ b1,
                            const float* __restrict__ wsd,
                            half_t* __restrict__ hx) {   // [NTOT,256] f16
    __shared__ float    sx[NN];
    __shared__ float    sas[NN][4], sad[NN][4];
    __shared__ unsigned smax[NN][4];
    __shared__ float    snum[NN][4], sden[NN][4];
    __shared__ float    sW1[256], sb1[256], sws[8];

    const int g = blockIdx.x, tid = threadIdx.x;
    if (tid < 8)   sws[tid] = wsd[tid];
    if (tid < 256) { sW1[tid] = W1[tid]; sb1[tid] = b1[tid]; }
    __syncthreads();

    if (tid < NN * 4) {
        int n = tid >> 2, h = tid & 3;
        float xv = x[g * NN + n];
        if (h == 0) sx[n] = xv;
        sas[n][h] = xv * sws[h];
        sad[n][h] = xv * sws[4 + h];
        smax[n][h] = 0u;
        snum[n][h] = 0.f;
        sden[n][h] = 0.f;
    }
    __syncthreads();

    // pass 1: per-(dst,head) max of leaky_relu(logit)
    for (int e = tid; e < EG + NN; e += blockDim.x) {
        int s, d;
        if (e < EG) { s = src[g * EG + e] - g * NN; d = dst[g * EG + e] - g * NN; }
        else        { s = d = e - EG; }                        // self loop
#pragma unroll
        for (int h = 0; h < 4; ++h) {
            float v = lrelu(sas[s][h] + sad[d][h]);
            atomicMax(&smax[d][h], f2ord(v));
        }
    }
    __syncthreads();

    // pass 2: w = exp(v - m); denom += w; num += w * x[src]
    for (int e = tid; e < EG + NN; e += blockDim.x) {
        int s, d;
        if (e < EG) { s = src[g * EG + e] - g * NN; d = dst[g * EG + e] - g * NN; }
        else        { s = d = e - EG; }
#pragma unroll
        for (int h = 0; h < 4; ++h) {
            float v = lrelu(sas[s][h] + sad[d][h]);
            float w = __expf(v - ord2f(smax[d][h]));
            atomicAdd(&sden[d][h], w);
            atomicAdd(&snum[d][h], w * sx[s]);
        }
    }
    __syncthreads();

    // out[n, h*64+c] = relu(t[n,h] * W1[h*64+c] + b1)  -> f16 for WMMA
    for (int i = tid; i < NN * 256; i += blockDim.x) {
        int n = i >> 8, j = i & 255, h = j >> 6;
        float t = snum[n][h] / sden[n][h];
        float v = relu_(t * sW1[j] + sb1[j]);
        hx[(size_t)(g * NN + n) * 256 + j] = (half_t)v;
    }
}

// =====================================================================
// WMMA f16 GEMM:  C[M,N] = A[M,K] * BT[N,K]^T  (+bias, relu), f32 accum
// 32x16 C tile per wave (2 accumulators share one B fragment).
// K is a template constant: the K-loop fully unrolls, so fragment
// double-buffering happens via SSA renaming (no register rotation) and
// the scheduler hoists each step's b128 loads above earlier WMMAs.
// =====================================================================
union Frag { v16h v; v8h h[2]; };

template <int K>
__global__ void gemm_wmma_kernel(const half_t* __restrict__ A,
                                 const half_t* __restrict__ BT,
                                 const float* __restrict__ bias,
                                 float* __restrict__ Cf,
                                 half_t* __restrict__ Ch,
                                 int M, int N, int do_relu) {
    const int lane    = threadIdx.x & 31;
    const int wave    = threadIdx.x >> 5;
    const int half_id = lane >> 4;          // K-half selector (ISA 16-bit A/B layout)
    const int r       = lane & 15;          // row of A / col of B within tile
    const int tileM   = blockIdx.x * 4 + wave;   // 32-row C tile per wave
    const int tileN   = blockIdx.y;
    if (tileM * 32 >= M) return;

    const half_t* Arow0 = A  + (size_t)(tileM * 32 + r)      * K + half_id * 8;
    const half_t* Arow1 = A  + (size_t)(tileM * 32 + 16 + r) * K + half_id * 8;
    const half_t* Brow  = BT + (size_t)(tileN * 16 + r)      * K + half_id * 8;

    v8f c0 = {}, c1 = {};
#pragma unroll
    for (int k0 = 0; k0 < K; k0 += 32) {
        Frag a0, a1, b;
        a0.h[0] = *(const v8h*)(Arow0 + k0);
        a0.h[1] = *(const v8h*)(Arow0 + k0 + 16);
        a1.h[0] = *(const v8h*)(Arow1 + k0);
        a1.h[1] = *(const v8h*)(Arow1 + k0 + 16);
        b.h[0]  = *(const v8h*)(Brow  + k0);
        b.h[1]  = *(const v8h*)(Brow  + k0 + 16);
        c0 = __builtin_amdgcn_wmma_f32_16x16x32_f16(
                 false, a0.v, false, b.v, (short)0, c0, false, false);
        c1 = __builtin_amdgcn_wmma_f32_16x16x32_f16(
                 false, a1.v, false, b.v, (short)0, c1, false, false);
    }

    const int col = tileN * 16 + r;
    const float bv = bias ? bias[col] : 0.f;
#pragma unroll
    for (int j = 0; j < 8; ++j) {           // C VGPR j: M = j + 8*half_id, N = r
        {
            int row = tileM * 32 + j + 8 * half_id;
            float v = c0[j] + bv;
            if (do_relu) v = relu_(v);
            Cf[(size_t)row * N + col] = v;
            if (Ch) Ch[(size_t)row * N + col] = (half_t)v;
        }
        {
            int row = tileM * 32 + 16 + j + 8 * half_id;
            float v = c1[j] + bv;
            if (do_relu) v = relu_(v);
            Cf[(size_t)row * N + col] = v;
            if (Ch) Ch[(size_t)row * N + col] = (half_t)v;
        }
    }
}

// =====================================================================
// GAT layer 2: one graph per block; h2 tile + accumulators live in LDS.
// Aggregation pass is wave-per-edge: 32 lanes spread the 128 channels.
// =====================================================================
__global__ void gat2_kernel(const float* __restrict__ h2,     // [NTOT,128]
                            const int* __restrict__ src,
                            const int* __restrict__ dst,
                            const float* __restrict__ a_src2, // [4,32]
                            const float* __restrict__ a_dst2,
                            const float* __restrict__ b2,     // [32]
                            half_t* __restrict__ gh) {        // [NTOT,32] f16
    __shared__ float    sh2[NN][128];
    __shared__ float    sout[NN][128];
    __shared__ float    sas[NN][4], sad[NN][4];
    __shared__ unsigned smax[NN][4];
    __shared__ float    sden[NN][4];
    __shared__ float    sa_s[128], sa_d[128];

    const int g = blockIdx.x, tid = threadIdx.x;
    if (tid < 128) { sa_s[tid] = a_src2[tid]; sa_d[tid] = a_dst2[tid]; }
    for (int i = tid; i < NN * 128; i += blockDim.x) {
        int n = i >> 7, j = i & 127;
        sh2[n][j]  = h2[(size_t)(g * NN + n) * 128 + j];
        sout[n][j] = 0.f;
    }
    if (tid < NN * 4) { int n = tid >> 2, h = tid & 3; smax[n][h] = 0u; sden[n][h] = 0.f; }
    __syncthreads();

    if (tid < NN * 4) {
        int n = tid >> 2, h = tid & 3;
        float s = 0.f, d = 0.f;
        for (int c = 0; c < 32; ++c) {
            float hv = sh2[n][h * 32 + c];
            s += hv * sa_s[h * 32 + c];
            d += hv * sa_d[h * 32 + c];
        }
        sas[n][h] = s; sad[n][h] = d;
    }
    __syncthreads();

    // pass 1: per-(dst,head) max
    for (int e = tid; e < EG + NN; e += blockDim.x) {
        int s, d;
        if (e < EG) { s = src[g * EG + e] - g * NN; d = dst[g * EG + e] - g * NN; }
        else        { s = d = e - EG; }
#pragma unroll
        for (int h = 0; h < 4; ++h) {
            float v = lrelu(sas[s][h] + sad[d][h]);
            atomicMax(&smax[d][h], f2ord(v));
        }
    }
    __syncthreads();

    // pass 2: wave-per-edge weighted aggregation (lane = channel within head)
    {
        const int wv = tid >> 5, ln = tid & 31;        // 8 waves x 32 lanes
        for (int e = wv; e < EG + NN; e += 8) {
            int s, d;
            if (e < EG) { s = src[g * EG + e] - g * NN; d = dst[g * EG + e] - g * NN; }
            else        { s = d = e - EG; }
#pragma unroll
            for (int h = 0; h < 4; ++h) {
                float v = lrelu(sas[s][h] + sad[d][h]);
                float w = __expf(v - ord2f(smax[d][h]));
                if (ln == 0) atomicAdd(&sden[d][h], w);
                atomicAdd(&sout[d][h * 32 + ln], w * sh2[s][h * 32 + ln]);
            }
        }
    }
    __syncthreads();

    // mean over heads (+b2, relu) -> f16 rows of g[2048,1056]
    for (int i = tid; i < NN * 32; i += blockDim.x) {
        int n = i >> 5, c = i & 31;
        float acc = 0.f;
#pragma unroll
        for (int h = 0; h < 4; ++h) acc += sout[n][h * 32 + c] / sden[n][h];
        float v = relu_(0.25f * acc + b2[c]);
        gh[(size_t)(g * NN + n) * 32 + c] = (half_t)v;
    }
}

// =====================================================================
// MLP head final layer: out[b] = sigmoid(m2[b,:]·lW3 + lb3)
// =====================================================================
__global__ void head_kernel(const float* __restrict__ m2,   // [2048,64]
                            const float* __restrict__ lW3,  // [64]
                            const float* __restrict__ lb3,
                            float* __restrict__ out) {
    int b = blockIdx.x * blockDim.x + threadIdx.x;
    if (b < NG) {
        float acc = lb3[0];
        for (int k = 0; k < 64; ++k) acc += m2[b * 64 + k] * lW3[k];
        out[b] = 1.f / (1.f + __expf(-acc));
    }
}

// =====================================================================
extern "C" void kernel_launch(void* const* d_in, const int* in_sizes, int n_in,
                              void* d_out, int out_size, void* d_ws, size_t ws_size,
                              hipStream_t stream) {
    const float* x      = (const float*)d_in[0];
    const int*   ei     = (const int*)  d_in[1];
    const float* W1     = (const float*)d_in[2];
    const float* a_src1 = (const float*)d_in[3];
    const float* a_dst1 = (const float*)d_in[4];
    const float* b1     = (const float*)d_in[5];
    const float* W2     = (const float*)d_in[6];
    const float* a_src2 = (const float*)d_in[7];
    const float* a_dst2 = (const float*)d_in[8];
    const float* b2     = (const float*)d_in[9];
    const float* lW1    = (const float*)d_in[10];
    const float* lb1    = (const float*)d_in[11];
    const float* lW2    = (const float*)d_in[12];
    const float* lb2    = (const float*)d_in[13];
    const float* lW3    = (const float*)d_in[14];
    const float* lb3    = (const float*)d_in[15];
    const int*   srcp   = ei;
    const int*   dstp   = ei + ETOT;

    // workspace carve-out (256B aligned)
    char*  w   = (char*)d_ws;
    size_t off = 0;
    auto alloc = [&](size_t bytes) -> void* {
        void* p = w + off;
        off = (off + bytes + 255) & ~size_t(255);
        return p;
    };
    float*  wsd  = (float*) alloc(8 * sizeof(float));
    half_t* hx   = (half_t*)alloc((size_t)NTOT * 256 * sizeof(half_t)); // 34.6 MB
    half_t* w2t  = (half_t*)alloc((size_t)128 * 256 * sizeof(half_t));
    float*  h2f  = (float*) alloc((size_t)NTOT * 128 * sizeof(float));  // 34.6 MB
    half_t* gh   = (half_t*)alloc((size_t)NTOT * 32 * sizeof(half_t));  // 4.3 MB
    half_t* lw1t = (half_t*)alloc((size_t)256 * 1056 * sizeof(half_t));
    float*  m1f  = (float*) alloc((size_t)NG * 256 * sizeof(float));
    half_t* m1h  = (half_t*)alloc((size_t)NG * 256 * sizeof(half_t));
    half_t* lw2t = (half_t*)alloc((size_t)64 * 256 * sizeof(half_t));
    float*  m2f  = (float*) alloc((size_t)NG * 64 * sizeof(float));
    (void)ws_size; (void)in_sizes; (void)n_in; (void)out_size;

    // --- prep: alpha dots + f16 weight transposes ---
    prep_alpha1_kernel<<<1, 64, 0, stream>>>(W1, a_src1, a_dst1, wsd);
    transpose_f16_kernel<<<(256 * 128 + 255) / 256, 256, 0, stream>>>(W2, w2t, 256, 128);
    transpose_f16_kernel<<<(1056 * 256 + 255) / 256, 256, 0, stream>>>(lW1, lw1t, 1056, 256);
    transpose_f16_kernel<<<(256 * 64 + 255) / 256, 256, 0, stream>>>(lW2, lw2t, 256, 64);

    // --- GAT layer 1 (factorized) -> hx f16 [NTOT,256] ---
    gat1_kernel<<<NG, 256, 0, stream>>>(x, srcp, dstp, W1, b1, wsd, hx);

    // --- h2 = hx @ W2   [67584,256]x[256,128] (WMMA, K=256) ---
    {
        dim3 grid((NTOT / 32 + 3) / 4, 128 / 16);
        gemm_wmma_kernel<256><<<grid, 128, 0, stream>>>(hx, w2t, nullptr, h2f, nullptr,
                                                        NTOT, 128, 0);
    }

    // --- GAT layer 2 -> g f16 [2048,1056] ---
    gat2_kernel<<<NG, 256, 0, stream>>>(h2f, srcp, dstp, a_src2, a_dst2, b2, gh);

    // --- MLP: m1 = relu(g @ lW1 + lb1)   [2048,1056]x[1056,256] (WMMA, K=1056) ---
    {
        dim3 grid((NG / 32 + 3) / 4, 256 / 16);
        gemm_wmma_kernel<1056><<<grid, 128, 0, stream>>>(gh, lw1t, lb1, m1f, m1h,
                                                         NG, 256, 1);
    }
    // --- m2 = relu(m1 @ lW2 + lb2)   [2048,256]x[256,64] (WMMA, K=256) ---
    {
        dim3 grid((NG / 32 + 3) / 4, 64 / 16);
        gemm_wmma_kernel<256><<<grid, 128, 0, stream>>>(m1h, lw2t, lb2, m2f, nullptr,
                                                        NG, 64, 1);
    }
    // --- sigmoid head ---
    head_kernel<<<(NG + 255) / 256, 256, 0, stream>>>(m2f, lW3, lb3, (float*)d_out);
}
